// FlashMHACustomMasking_2645699854767
// MI455X (gfx1250) — matrix operand order repaired and checked by reading, not compiled, not run on hardware
//
#include <hip/hip_runtime.h>
#include <hip/hip_bf16.h>

typedef __attribute__((ext_vector_type(8)))  __bf16 v8bf;
typedef __attribute__((ext_vector_type(16))) __bf16 v16bf;
typedef __attribute__((ext_vector_type(8)))  float  v8f;

union V16U { v16bf v; v8bf h8[2]; };

#define NB  2
#define NS  2048
#define NH  16
#define ND  64
#define DM  1024
#define SCALE 0.125f

__device__ __forceinline__ __bf16 f2bf(float f) {
    unsigned int u = __builtin_bit_cast(unsigned int, f);
    unsigned int r = (u + 0x7FFFu + ((u >> 16) & 1u)) >> 16;
    unsigned short s = (unsigned short)r;
    return __builtin_bit_cast(__bf16, s);
}

// reduce across the 16 lanes of this lane's half (xor masks 1,2,4,8 stay in-half)
__device__ __forceinline__ float red_max16(float x) {
    int v, o;
    v = __builtin_bit_cast(int, x);
    o = __builtin_amdgcn_ds_swizzle(v, (1 << 10) | 0x1f); x = fmaxf(x, __builtin_bit_cast(float, o));
    v = __builtin_bit_cast(int, x);
    o = __builtin_amdgcn_ds_swizzle(v, (2 << 10) | 0x1f); x = fmaxf(x, __builtin_bit_cast(float, o));
    v = __builtin_bit_cast(int, x);
    o = __builtin_amdgcn_ds_swizzle(v, (4 << 10) | 0x1f); x = fmaxf(x, __builtin_bit_cast(float, o));
    v = __builtin_bit_cast(int, x);
    o = __builtin_amdgcn_ds_swizzle(v, (8 << 10) | 0x1f); x = fmaxf(x, __builtin_bit_cast(float, o));
    return x;
}

__device__ __forceinline__ float red_sum16(float x) {
    int v, o;
    v = __builtin_bit_cast(int, x);
    o = __builtin_amdgcn_ds_swizzle(v, (1 << 10) | 0x1f); x += __builtin_bit_cast(float, o);
    v = __builtin_bit_cast(int, x);
    o = __builtin_amdgcn_ds_swizzle(v, (2 << 10) | 0x1f); x += __builtin_bit_cast(float, o);
    v = __builtin_bit_cast(int, x);
    o = __builtin_amdgcn_ds_swizzle(v, (4 << 10) | 0x1f); x += __builtin_bit_cast(float, o);
    v = __builtin_bit_cast(int, x);
    o = __builtin_amdgcn_ds_swizzle(v, (8 << 10) | 0x1f); x += __builtin_bit_cast(float, o);
    return x;
}

// ---------------- conversion kernels ----------------

// q,k,v: (B,S,H,D) fp32 -> Qb,Kb: bf16 [B,H,S,D]; Vt: bf16 [B,H,D,S]
__global__ __launch_bounds__(256) void cvt_qkv_kernel(
    const float* __restrict__ q, const float* __restrict__ k, const float* __restrict__ v,
    __bf16* __restrict__ Qb, __bf16* __restrict__ Kb, __bf16* __restrict__ Vt)
{
    int idx = blockIdx.x * 256 + threadIdx.x;       // 0 .. B*S*H*D-1
    int d  = idx & (ND - 1);
    int hh = (idx >> 6) & (NH - 1);
    int s  = (idx >> 10) & (NS - 1);
    int bb = idx >> 21;
    size_t hm = (size_t)(bb * NH + hh);
    size_t dsti = (hm * NS + s) * ND + d;
    // q/k/v fp32 are read once -> non-temporal
    Qb[dsti] = f2bf(__builtin_nontemporal_load(q + idx));
    Kb[dsti] = f2bf(__builtin_nontemporal_load(k + idx));
    Vt[(hm * ND + d) * NS + s] = f2bf(__builtin_nontemporal_load(v + idx));
}

__global__ __launch_bounds__(256) void cvt_w_kernel(
    const float* __restrict__ W, __bf16* __restrict__ Wb)
{
    int idx = blockIdx.x * 256 + threadIdx.x;
    Wb[idx] = f2bf(__builtin_nontemporal_load(W + idx));
}

// ---------------- flash attention ----------------
// 1 wave = one 16-query tile of one (b,h). grid = B*H*(S/16)/4 blocks of 128 threads.
__global__ __launch_bounds__(128) void attn_kernel(
    const __bf16* __restrict__ Qb, const __bf16* __restrict__ Kb,
    const __bf16* __restrict__ Vt, const float* __restrict__ mask,
    __bf16* __restrict__ ctx)
{
    __shared__ __align__(32) __bf16 Plds[4][16 * 32];

    int lane = threadIdx.x & 31;
    int wid  = threadIdx.x >> 5;
    int tile = blockIdx.x * 4 + wid;            // 0..4095
    int b   = tile >> 11;                       // /(H * S/16)
    int rem = tile & 2047;
    int h   = rem >> 7;
    int q0  = (rem & 127) << 4;

    int n   = lane & 15;
    int hi  = lane >> 4;
    int klo = hi ? 8 : 0;

    const __bf16* qbase = Qb + (size_t)(b * NH + h) * NS * ND;
    const __bf16* kbase = Kb + (size_t)(b * NH + h) * NS * ND;
    const __bf16* vbase = Vt + (size_t)(b * NH + h) * ND * NS;
    const float*  mbase = mask + (size_t)h * NS * NS;

    // Q A-operands (rows q0..q0+15, D split into two K=32 blocks)
    V16U a0, a1;
    {
        const __bf16* qr = qbase + (size_t)(q0 + n) * ND;
        a0.h8[0] = *(const v8bf*)(qr + hi * 16);
        a0.h8[1] = *(const v8bf*)(qr + hi * 16 + 8);
        a1.h8[0] = *(const v8bf*)(qr + 32 + hi * 16);
        a1.h8[1] = *(const v8bf*)(qr + 32 + hi * 16 + 8);
    }

    v8f acc0 = {}, acc1 = {}, acc2 = {}, acc3 = {};
    float mrun[8], lrun[8];
#pragma unroll
    for (int r = 0; r < 8; r++) { mrun[r] = -3.0e38f; lrun[r] = 0.0f; }

    __bf16* pl = &Plds[wid][0];

    for (int kb = 0; kb < NS; kb += 32) {
        // ---- S = Q K^T over this 32-key block (two 16-key subtiles) ----
        const __bf16* k0p = kbase + (size_t)(kb + n) * ND + hi * 16;
        const __bf16* k1p = kbase + (size_t)(kb + 16 + n) * ND + hi * 16;
        v16bf b00 = *(const v16bf*)(k0p);
        v16bf b01 = *(const v16bf*)(k0p + 32);
        v16bf b10 = *(const v16bf*)(k1p);
        v16bf b11 = *(const v16bf*)(k1p + 32);

        v8f z = {};
        v8f c0 = __builtin_amdgcn_wmma_f32_16x16x32_bf16(false, a0.v, false, b00, (short)0, z,  false, false);
        c0     = __builtin_amdgcn_wmma_f32_16x16x32_bf16(false, a1.v, false, b01, (short)0, c0, false, false);
        v8f c1 = __builtin_amdgcn_wmma_f32_16x16x32_bf16(false, a0.v, false, b10, (short)0, z,  false, false);
        c1     = __builtin_amdgcn_wmma_f32_16x16x32_bf16(false, a1.v, false, b11, (short)0, c1, false, false);

        // ---- scale + additive mask (268 MB streamed once -> non-temporal) ----
        float s0[8], s1[8];
#pragma unroll
        for (int r = 0; r < 8; r++) {
            const float* mrow = mbase + (size_t)(q0 + r + 8 * hi) * NS + kb;
            s0[r] = c0[r] * SCALE + __builtin_nontemporal_load(mrow + n);
            s1[r] = c1[r] * SCALE + __builtin_nontemporal_load(mrow + 16 + n);
        }

        // ---- online softmax ----
        float p0[8], p1[8], alpha[8];
#pragma unroll
        for (int r = 0; r < 8; r++) {
            float bm = red_max16(fmaxf(s0[r], s1[r]));
            float nm = fmaxf(mrun[r], bm);
            alpha[r] = __expf(mrun[r] - nm);
            p0[r] = __expf(s0[r] - nm);
            p1[r] = __expf(s1[r] - nm);
            float bs = red_sum16(p0[r] + p1[r]);
            lrun[r] = lrun[r] * alpha[r] + bs;
            mrun[r] = nm;
        }
#pragma unroll
        for (int r = 0; r < 8; r++) {
            acc0[r] *= alpha[r]; acc1[r] *= alpha[r];
            acc2[r] *= alpha[r]; acc3[r] *= alpha[r];
        }

        // ---- stage P (16x32 bf16) in LDS: D-layout -> A-layout transpose ----
#pragma unroll
        for (int r = 0; r < 8; r++) {
            int row = r + 8 * hi;
            pl[row * 32 + n]      = f2bf(p0[r]);
            pl[row * 32 + 16 + n] = f2bf(p1[r]);
        }
        V16U ap;
        {
            const __bf16* pr = pl + n * 32;       // A row m = lane%16
            ap.h8[0] = *(const v8bf*)(pr + hi * 16);
            ap.h8[1] = *(const v8bf*)(pr + hi * 16 + 8);
        }

        // ---- O += P V (V^T layout makes B-operand contiguous) ----
        const __bf16* vb = vbase + (size_t)n * NS + kb + hi * 16;
        v16bf vb0 = *(const v16bf*)(vb);
        v16bf vb1 = *(const v16bf*)(vb + 16 * NS);
        v16bf vb2 = *(const v16bf*)(vb + 32 * NS);
        v16bf vb3 = *(const v16bf*)(vb + 48 * NS);
        acc0 = __builtin_amdgcn_wmma_f32_16x16x32_bf16(false, ap.v, false, vb0, (short)0, acc0, false, false);
        acc1 = __builtin_amdgcn_wmma_f32_16x16x32_bf16(false, ap.v, false, vb1, (short)0, acc1, false, false);
        acc2 = __builtin_amdgcn_wmma_f32_16x16x32_bf16(false, ap.v, false, vb2, (short)0, acc2, false, false);
        acc3 = __builtin_amdgcn_wmma_f32_16x16x32_bf16(false, ap.v, false, vb3, (short)0, acc3, false, false);
    }

    // ---- normalize and store ctx (bf16, [B,S,H*D]) ----
#pragma unroll
    for (int r = 0; r < 8; r++) {
        float inv = 1.0f / lrun[r];
        size_t base = (size_t)(b * NS + q0 + r + 8 * hi) * DM + h * ND + n;
        ctx[base]      = f2bf(acc0[r] * inv);
        ctx[base + 16] = f2bf(acc1[r] * inv);
        ctx[base + 32] = f2bf(acc2[r] * inv);
        ctx[base + 48] = f2bf(acc3[r] * inv);
    }
}

// ---------------- output projection: Y = ctx @ W^T + b ----------------
// 1 wave per 16x16 output tile. (B*S/16) x (DM/16) = 256 x 64 tiles.
__global__ __launch_bounds__(128) void proj_kernel(
    const __bf16* __restrict__ ctx, const __bf16* __restrict__ Wb,
    const float* __restrict__ bias, float* __restrict__ out)
{
    int lane = threadIdx.x & 31;
    int wid  = threadIdx.x >> 5;
    int tile = blockIdx.x * 4 + wid;     // 0..16383
    int i0 = (tile >> 6) << 4;
    int j0 = (tile & 63) << 4;

    int n   = lane & 15;
    int hi  = lane >> 4;
    int klo = hi ? 8 : 0;

    float bv = bias[j0 + n];
    v8f acc;
#pragma unroll
    for (int r = 0; r < 8; r++) acc[r] = bv;

    const __bf16* arow = ctx + (size_t)(i0 + n) * DM;
    const __bf16* brow = Wb + (size_t)(j0 + n) * DM + hi * 16;

    for (int cb = 0; cb < DM; cb += 32) {
        V16U a;
        a.h8[0] = *(const v8bf*)(arow + cb + hi * 16);
        a.h8[1] = *(const v8bf*)(arow + cb + hi * 16 + 8);
        v16bf bb = *(const v16bf*)(brow + cb);
        acc = __builtin_amdgcn_wmma_f32_16x16x32_bf16(false, a.v, false, bb, (short)0, acc, false, false);
    }

    // output written once, never re-read on device -> non-temporal stores
#pragma unroll
    for (int r = 0; r < 8; r++) {
        __builtin_nontemporal_store(acc[r], out + (size_t)(i0 + r + 8 * hi) * DM + j0 + n);
    }
}

// ---------------- host launcher ----------------
extern "C" void kernel_launch(void* const* d_in, const int* in_sizes, int n_in,
                              void* d_out, int out_size, void* d_ws, size_t ws_size,
                              hipStream_t stream) {
    const float* q    = (const float*)d_in[0];
    const float* k    = (const float*)d_in[1];
    const float* v    = (const float*)d_in[2];
    const float* mask = (const float*)d_in[3];
    const float* W    = (const float*)d_in[4];
    const float* bias = (const float*)d_in[5];
    float* out = (float*)d_out;

    const size_t QKV_BYTES = (size_t)NB * NH * NS * ND * sizeof(__bf16); // 8 MB each
    char* ws = (char*)d_ws;
    __bf16* Qb = (__bf16*)(ws);
    __bf16* Kb = (__bf16*)(ws + QKV_BYTES);
    __bf16* Vt = (__bf16*)(ws + 2 * QKV_BYTES);
    __bf16* Cx = (__bf16*)(ws + 3 * QKV_BYTES);
    __bf16* Wb = (__bf16*)(ws + 4 * QKV_BYTES);

    const int NQKV = NB * NS * NH * ND;          // 4194304
    cvt_qkv_kernel<<<NQKV / 256, 256, 0, stream>>>(q, k, v, Qb, Kb, Vt);
    cvt_w_kernel<<<(DM * DM) / 256, 256, 0, stream>>>(W, Wb);

    // B*H*(S/16) = 4096 query tiles, 4 waves per block
    attn_kernel<<<1024, 128, 0, stream>>>(Qb, Kb, Vt, mask, Cx);

    // 256*64 = 16384 output tiles, 4 waves per block
    proj_kernel<<<4096, 128, 0, stream>>>(Cx, Wb, bias, out);
}
